// ConvMambaEncoder_40785009443169
// MI455X (gfx1250) — matrix operand order repaired and checked by reading, not compiled, not run on hardware
//
#include <hip/hip_runtime.h>
#include <hip/hip_bf16.h>

#define BATCH 4
#define SEQ   2048
#define IN_DIM 16
#define D_MODEL 256
#define D_INNER 512
#define D_STATE 16
#define DT_RANK 16
#define OUT_DIM 128

typedef __attribute__((ext_vector_type(16))) __bf16 v16bf;
typedef __attribute__((ext_vector_type(8)))  __bf16 v8bf;
typedef __attribute__((ext_vector_type(8)))  float  v8f;

// ---------------------------------------------------------------------------
// WMMA fragment helpers (bf16 A/B, f32 C/D), wave32, blockDim == 32.
// A (16x32, row-major src, row = M): lane<16 -> K 0-7 & 16-23, lane>=16 -> +8.
// B (32x16 from N-major src, row = N): same K split as A (lane<16 -> K 0-7 & 16-23, lane>=16 -> +8).
// C/D: m = r + 8*(lane>>4), n = lane&15.
// ---------------------------------------------------------------------------
__device__ inline v16bf ld_frag_a(const __bf16* __restrict__ src, int ld, int row0, int k0) {
  int lane = threadIdx.x & 31;
  int r = row0 + (lane & 15);
  int kh = (lane >> 4) * 8;
  const v8bf* p = reinterpret_cast<const v8bf*>(src + (size_t)r * ld + k0 + kh);
  v8bf lo = p[0];
  v8bf hi = p[2];            // +16 elements
  v16bf f;
#pragma unroll
  for (int i = 0; i < 8; ++i) { f[i] = lo[i]; f[8 + i] = hi[i]; }
  return f;
}

__device__ inline v16bf ld_frag_a_clamped(const __bf16* __restrict__ src, int ld,
                                          int row0, int k0, int nrows) {
  int lane = threadIdx.x & 31;
  int r = row0 + (lane & 15);
  v16bf f;
  if (r >= 0 && r < nrows) {
    int kh = (lane >> 4) * 8;
    const v8bf* p = reinterpret_cast<const v8bf*>(src + (size_t)r * ld + k0 + kh);
    v8bf lo = p[0];
    v8bf hi = p[2];
#pragma unroll
    for (int i = 0; i < 8; ++i) { f[i] = lo[i]; f[8 + i] = hi[i]; }
  } else {
#pragma unroll
    for (int i = 0; i < 16; ++i) f[i] = (__bf16)0.f;
  }
  return f;
}

__device__ inline v16bf ld_frag_b(const __bf16* __restrict__ src, int ld, int n0, int k0) {
  int lane = threadIdx.x & 31;
  int n = n0 + (lane & 15);
  int kb = k0 + ((lane >> 4) << 3);
  const v8bf* p = reinterpret_cast<const v8bf*>(src + (size_t)n * ld + kb);
  v8bf lo = p[0];
  v8bf hi = p[2];
  v16bf f;
#pragma unroll
  for (int i = 0; i < 8; ++i) { f[i] = lo[i]; f[8 + i] = hi[i]; }
  return f;
}

__device__ inline void st_tile_f32(float* __restrict__ dst, int ld, int row0, int col0,
                                   v8f acc) {
  int lane = threadIdx.x & 31;
  int n = lane & 15, mb = (lane >> 4) * 8;
#pragma unroll
  for (int r = 0; r < 8; ++r)
    dst[(size_t)(row0 + mb + r) * ld + col0 + n] = acc[r];
}

#define WMMA_BF16(a, b, c) \
  __builtin_amdgcn_wmma_f32_16x16x32_bf16(false, (a), false, (b), (short)0, (c), false, false)

// ---------------------------------------------------------------------------
// Weight prep: fp32 -> bf16 (+ layout changes)
// ---------------------------------------------------------------------------
__global__ void k_cast_bf16(const float* __restrict__ src, __bf16* __restrict__ dst, int n) {
  int i = blockIdx.x * 256 + threadIdx.x;
  if (i < n) dst[i] = (__bf16)src[i];
}

// conv2_w (oc, ic, tap) -> w2b[oc][tap*256+ic]  (K = tap-major, 768 wide)
__global__ void k_prep_w2(const float* __restrict__ w, __bf16* __restrict__ w2b) {
  int i = blockIdx.x * 256 + threadIdx.x;
  if (i >= D_MODEL * 768) return;
  int oc = i / 768, k = i % 768, tap = k >> 8, ic = k & 255;
  w2b[i] = (__bf16)w[(oc * D_MODEL + ic) * 3 + tap];
}

// dt_proj_w (512,16) -> dtwb[512][32], K 16..31 zero-padded
__global__ void k_prep_dtw(const float* __restrict__ w, __bf16* __restrict__ dtwb) {
  int i = blockIdx.x * 256 + threadIdx.x;
  if (i >= D_INNER * 32) return;
  int d = i >> 5, k = i & 31;
  dtwb[i] = (k < DT_RANK) ? (__bf16)w[d * DT_RANK + k] : (__bf16)0.f;
}

// ---------------------------------------------------------------------------
// conv1 (16->256, k=5, pad 2) + relu, output bf16 (B,S,256)
// ---------------------------------------------------------------------------
__global__ void k_conv1(const float* __restrict__ x, const float* __restrict__ w,
                        const float* __restrict__ bias, __bf16* __restrict__ h1b) {
  __shared__ float xs[IN_DIM * 5];
  int b = blockIdx.x / SEQ;
  int s = blockIdx.x % SEQ;
  int t = threadIdx.x;
  if (t < IN_DIM * 5) {
    int ic = t / 5, tap = t % 5;
    int sp = s + tap - 2;
    xs[t] = (sp >= 0 && sp < SEQ) ? x[((size_t)b * IN_DIM + ic) * SEQ + sp] : 0.f;
  }
  __syncthreads();
  int d = t;                       // 0..255
  float acc = bias[d];
  const float* wd = w + (size_t)d * (IN_DIM * 5);
#pragma unroll
  for (int i = 0; i < IN_DIM * 5; ++i) acc += xs[i] * wd[i];
  acc = fmaxf(acc, 0.f);
  h1b[((size_t)b * SEQ + s) * D_MODEL + d] = (__bf16)acc;
}

// ---------------------------------------------------------------------------
// conv2 (256->256, k=3, pad 1) + relu as WMMA GEMM, K = 3*256 tap-major.
// One wave -> 16 seq x 64 oc (4 accumulators, A fragment reused 4x).
// Interior seq tiles skip the row-clamp path entirely (uniform branch).
// ---------------------------------------------------------------------------
__global__ void k_conv2(const __bf16* __restrict__ h1b, const __bf16* __restrict__ w2b,
                        const float* __restrict__ bias, __bf16* __restrict__ ub) {
  int s0 = blockIdx.x * 16, oc0 = blockIdx.y * 64, b = blockIdx.z;
  const __bf16* A = h1b + (size_t)b * SEQ * D_MODEL;
  v8f acc[4] = {};
  const bool interior = (s0 != 0) && (s0 != SEQ - 16);
  if (interior) {
    for (int kc = 0; kc < 768; kc += 32) {
      int tap = kc >> 8;
      int ic0 = kc & 255;
      v16bf af = ld_frag_a(A, D_MODEL, s0 + tap - 1, ic0);
#pragma unroll
      for (int j = 0; j < 4; ++j) {
        v16bf bf = ld_frag_b(w2b, 768, oc0 + j * 16, kc);
        acc[j] = WMMA_BF16(af, bf, acc[j]);
      }
    }
  } else {
    for (int kc = 0; kc < 768; kc += 32) {
      int tap = kc >> 8;
      int ic0 = kc & 255;
      v16bf af = ld_frag_a_clamped(A, D_MODEL, s0 + tap - 1, ic0, SEQ);
#pragma unroll
      for (int j = 0; j < 4; ++j) {
        v16bf bf = ld_frag_b(w2b, 768, oc0 + j * 16, kc);
        acc[j] = WMMA_BF16(af, bf, acc[j]);
      }
    }
  }
  int lane = threadIdx.x & 31;
  int n = lane & 15, mb = (lane >> 4) * 8;
#pragma unroll
  for (int j = 0; j < 4; ++j) {
    float bv = bias[oc0 + j * 16 + n];
#pragma unroll
    for (int r = 0; r < 8; ++r) {
      float v = fmaxf(acc[j][r] + bv, 0.f);
      ub[((size_t)b * SEQ + s0 + mb + r) * D_MODEL + oc0 + j * 16 + n] = (__bf16)v;
    }
  }
}

// ---------------------------------------------------------------------------
// in_proj: (B,S,256)bf16 @ (1024,256)bf16^T -> xz fp32 (B,S,1024)
// One wave -> 16 seq x 64 out (4 accumulators), prefetch next-K A rows.
// ---------------------------------------------------------------------------
__global__ void k_inproj(const __bf16* __restrict__ ub, const __bf16* __restrict__ wpb,
                         float* __restrict__ xz) {
  int s0 = blockIdx.x * 16, j0 = blockIdx.y * 64, b = blockIdx.z;
  const __bf16* A = ub + (size_t)b * SEQ * D_MODEL;
  v8f acc[4] = {};
  int lane = threadIdx.x & 31;
  for (int k = 0; k < D_MODEL; k += 32) {
    v16bf af = ld_frag_a(A, D_MODEL, s0, k);
    if (k + 32 < D_MODEL)
      __builtin_prefetch(A + (size_t)(s0 + (lane & 15)) * D_MODEL + k + 32, 0, 1);
#pragma unroll
    for (int j = 0; j < 4; ++j) {
      v16bf bf = ld_frag_b(wpb, D_MODEL, j0 + j * 16, k);
      acc[j] = WMMA_BF16(af, bf, acc[j]);
    }
  }
  float* dst = xz + (size_t)b * SEQ * 1024;
#pragma unroll
  for (int j = 0; j < 4; ++j) st_tile_f32(dst, 1024, s0, j0 + j * 16, acc[j]);
}

// ---------------------------------------------------------------------------
// depthwise causal conv (k=4) + silu on xm = xz[..,0:512]
// ---------------------------------------------------------------------------
__global__ void k_dwconv(const float* __restrict__ xz, const float* __restrict__ dww,
                         const float* __restrict__ dwb, float* __restrict__ xc,
                         __bf16* __restrict__ xcb) {
  int b = blockIdx.x / SEQ, s = blockIdx.x % SEQ, d = threadIdx.x;
  float acc = dwb[d];
#pragma unroll
  for (int t = 0; t < 4; ++t) {
    int sp = s - 3 + t;
    if (sp >= 0) acc += xz[((size_t)b * SEQ + sp) * 1024 + d] * dww[d * 4 + t];
  }
  float v = acc / (1.f + __expf(-acc));   // silu
  size_t o = ((size_t)b * SEQ + s) * D_INNER + d;
  xc[o] = v;
  xcb[o] = (__bf16)v;
}

// ---------------------------------------------------------------------------
// x_proj: (B,S,512)bf16 @ (48,512)^T -> dbc fp32 (B,S,48); all 3 N-tiles in
// one wave (A fragment reused 3x). Tile 0 also written bf16 for dt_proj.
// ---------------------------------------------------------------------------
__global__ void k_xproj(const __bf16* __restrict__ xcb, const __bf16* __restrict__ xpb,
                        float* __restrict__ dbc, __bf16* __restrict__ dtinb) {
  int s0 = blockIdx.x * 16, b = blockIdx.y;
  const __bf16* A = xcb + (size_t)b * SEQ * D_INNER;
  v8f acc[3] = {};
  for (int k = 0; k < D_INNER; k += 32) {
    v16bf af = ld_frag_a(A, D_INNER, s0, k);
#pragma unroll
    for (int j = 0; j < 3; ++j) {
      v16bf bf = ld_frag_b(xpb, D_INNER, j * 16, k);
      acc[j] = WMMA_BF16(af, bf, acc[j]);
    }
  }
  int lane = threadIdx.x & 31;
  int n = lane & 15, mb = (lane >> 4) * 8;
#pragma unroll
  for (int j = 0; j < 3; ++j) {
#pragma unroll
    for (int r = 0; r < 8; ++r) {
      int s = s0 + mb + r;
      float v = acc[j][r];
      dbc[((size_t)b * SEQ + s) * 48 + j * 16 + n] = v;
      if (j == 0) dtinb[((size_t)b * SEQ + s) * DT_RANK + n] = (__bf16)v;
    }
  }
}

// ---------------------------------------------------------------------------
// dt_proj (K=16 padded to 32) + softplus -> dt fp32 (B,S,512)
// One wave -> 16 seq x 64 d (A loaded once, 4 WMMAs).
// ---------------------------------------------------------------------------
__global__ void k_dtproj(const __bf16* __restrict__ dtinb, const __bf16* __restrict__ dtwb,
                         const float* __restrict__ dtb, float* __restrict__ dt) {
  int s0 = blockIdx.x * 16, d0 = blockIdx.y * 64, b = blockIdx.z;
  int lane = threadIdx.x & 31;
  int r = s0 + (lane & 15);
  int kh = (lane >> 4) * 8;
  const __bf16* p = dtinb + ((size_t)b * SEQ + r) * DT_RANK + kh;
  v8bf lo = *reinterpret_cast<const v8bf*>(p);
  v16bf a;
#pragma unroll
  for (int i = 0; i < 8; ++i) { a[i] = lo[i]; a[8 + i] = (__bf16)0.f; }  // K>=16 pad
  v8f acc[4] = {};
#pragma unroll
  for (int j = 0; j < 4; ++j) {
    v16bf bf = ld_frag_b(dtwb, 32, d0 + j * 16, 0);
    acc[j] = WMMA_BF16(a, bf, acc[j]);
  }
  int n = lane & 15, mb = (lane >> 4) * 8;
#pragma unroll
  for (int j = 0; j < 4; ++j) {
    float bias = dtb[d0 + j * 16 + n];
#pragma unroll
    for (int rr = 0; rr < 8; ++rr) {
      float v = acc[j][rr] + bias;
      v = (v > 20.f) ? v : log1pf(__expf(v));   // softplus
      dt[((size_t)b * SEQ + s0 + mb + rr) * D_INNER + d0 + j * 16 + n] = v;
    }
  }
}

// ---------------------------------------------------------------------------
// Selective scan: one block per batch, one lane per d_inner channel.
// States in registers; B/C staged in LDS. Also folds the sequence mean:
// ymean[b,d] = mean_s y[b,s,d]  (out_proj commutes with the mean).
// ---------------------------------------------------------------------------
__global__ void k_scan(const float* __restrict__ dt, const float* __restrict__ xc,
                       const float* __restrict__ xz, const float* __restrict__ dbc,
                       const float* __restrict__ A_log, const float* __restrict__ D_skip,
                       float* __restrict__ ymean) {
  __shared__ float bc[2 * D_STATE];
  int b = blockIdx.x, d = threadIdx.x;        // d in [0,512)
  float a[D_STATE], h[D_STATE];
#pragma unroll
  for (int n = 0; n < D_STATE; ++n) {
    a[n] = -__expf(A_log[d * D_STATE + n]);
    h[n] = 0.f;
  }
  float Dk = D_skip[d];
  float acc = 0.f;
  for (int s = 0; s < SEQ; ++s) {
    if (d < 2 * D_STATE) bc[d] = dbc[((size_t)b * SEQ + s) * 48 + DT_RANK + d];
    __syncthreads();
    size_t o = ((size_t)b * SEQ + s) * D_INNER + d;
    float dtv = dt[o];
    float xv = xc[o];
    float zv = xz[((size_t)b * SEQ + s) * 1024 + D_INNER + d];
    float y = 0.f;
#pragma unroll
    for (int n = 0; n < D_STATE; ++n) {
      h[n] = __expf(dtv * a[n]) * h[n] + dtv * bc[n] * xv;
      y += h[n] * bc[D_STATE + n];
    }
    y += xv * Dk;
    y *= zv / (1.f + __expf(-zv));            // * silu(z)
    acc += y;
    __syncthreads();
  }
  ymean[b * D_INNER + d] = acc * (1.f / SEQ);
}

// pooled = ymean @ out_proj_w^T   (4x512 @ 512x256)
__global__ void k_pool(const float* __restrict__ ymean, const float* __restrict__ opw,
                       float* __restrict__ pooled) {
  int b = blockIdx.x, dm = threadIdx.x;       // 256
  float acc = 0.f;
  for (int k = 0; k < D_INNER; ++k) acc += ymean[b * D_INNER + k] * opw[dm * D_INNER + k];
  pooled[b * D_MODEL + dm] = acc;
}

// out = pooled @ proj_w^T + proj_b   (4x256 @ 256x128)
__global__ void k_final(const float* __restrict__ pooled, const float* __restrict__ pw,
                        const float* __restrict__ pb, float* __restrict__ out) {
  int b = blockIdx.x, o = threadIdx.x;        // 128
  float acc = pb[o];
  for (int k = 0; k < D_MODEL; ++k) acc += pooled[b * D_MODEL + k] * pw[o * D_MODEL + k];
  out[b * OUT_DIM + o] = acc;
}

// ---------------------------------------------------------------------------
extern "C" void kernel_launch(void* const* d_in, const int* in_sizes, int n_in,
                              void* d_out, int out_size, void* d_ws, size_t ws_size,
                              hipStream_t stream) {
  const float* x        = (const float*)d_in[0];
  const float* conv1_w  = (const float*)d_in[1];
  const float* conv1_b  = (const float*)d_in[2];
  const float* conv2_w  = (const float*)d_in[3];
  const float* conv2_b  = (const float*)d_in[4];
  const float* inproj_w = (const float*)d_in[5];
  const float* dw_w     = (const float*)d_in[6];
  const float* dw_b     = (const float*)d_in[7];
  const float* xproj_w  = (const float*)d_in[8];
  const float* dtproj_w = (const float*)d_in[9];
  const float* dtproj_b = (const float*)d_in[10];
  const float* A_log    = (const float*)d_in[11];
  const float* D_skip   = (const float*)d_in[12];
  const float* outproj_w= (const float*)d_in[13];
  const float* proj_w   = (const float*)d_in[14];
  const float* proj_b   = (const float*)d_in[15];
  float* out = (float*)d_out;

  char* ws = (char*)d_ws;
  size_t off = 0;
  auto alloc = [&](size_t bytes) -> void* {
    off = (off + 255) & ~(size_t)255;
    void* p = ws + off;
    off += bytes;
    return p;
  };
  __bf16* h1b   = (__bf16*)alloc((size_t)BATCH * SEQ * D_MODEL * 2);
  __bf16* ub    = (__bf16*)alloc((size_t)BATCH * SEQ * D_MODEL * 2);
  __bf16* w2b   = (__bf16*)alloc((size_t)D_MODEL * 768 * 2);
  __bf16* wpb   = (__bf16*)alloc((size_t)1024 * D_MODEL * 2);
  __bf16* xpb   = (__bf16*)alloc((size_t)48 * D_INNER * 2);
  __bf16* dtwb  = (__bf16*)alloc((size_t)D_INNER * 32 * 2);
  float*  xz    = (float*) alloc((size_t)BATCH * SEQ * 1024 * 4);
  float*  xc    = (float*) alloc((size_t)BATCH * SEQ * D_INNER * 4);
  __bf16* xcb   = (__bf16*)alloc((size_t)BATCH * SEQ * D_INNER * 2);
  float*  dbc   = (float*) alloc((size_t)BATCH * SEQ * 48 * 4);
  __bf16* dtinb = (__bf16*)alloc((size_t)BATCH * SEQ * DT_RANK * 2);
  float*  dt    = (float*) alloc((size_t)BATCH * SEQ * D_INNER * 4);
  float*  ymean = (float*) alloc((size_t)BATCH * D_INNER * 4);
  float*  pooled= (float*) alloc((size_t)BATCH * D_MODEL * 4);
  (void)ws_size; (void)in_sizes; (void)n_in; (void)out_size;

  // weight prep
  k_prep_w2 <<<(D_MODEL * 768 + 255) / 256, 256, 0, stream>>>(conv2_w, w2b);
  k_cast_bf16<<<(1024 * D_MODEL + 255) / 256, 256, 0, stream>>>(inproj_w, wpb, 1024 * D_MODEL);
  k_cast_bf16<<<(48 * D_INNER + 255) / 256, 256, 0, stream>>>(xproj_w, xpb, 48 * D_INNER);
  k_prep_dtw<<<(D_INNER * 32 + 255) / 256, 256, 0, stream>>>(dtproj_w, dtwb);

  // pipeline
  k_conv1 <<<dim3(BATCH * SEQ), 256, 0, stream>>>(x, conv1_w, conv1_b, h1b);
  k_conv2 <<<dim3(SEQ / 16, D_MODEL / 64, BATCH), 32, 0, stream>>>(h1b, w2b, conv2_b, ub);
  k_inproj<<<dim3(SEQ / 16, 1024 / 64, BATCH), 32, 0, stream>>>(ub, wpb, xz);
  k_dwconv<<<dim3(BATCH * SEQ), D_INNER, 0, stream>>>(xz, dw_w, dw_b, xc, xcb);
  k_xproj <<<dim3(SEQ / 16, BATCH), 32, 0, stream>>>(xcb, xpb, dbc, dtinb);
  k_dtproj<<<dim3(SEQ / 16, D_INNER / 64, BATCH), 32, 0, stream>>>(dtinb, dtwb, dtproj_b, dt);
  k_scan  <<<dim3(BATCH), D_INNER, 0, stream>>>(dt, xc, xz, dbc, A_log, D_skip, ymean);
  k_pool  <<<dim3(BATCH), D_MODEL, 0, stream>>>(ymean, outproj_w, pooled);
  k_final <<<dim3(BATCH), OUT_DIM, 0, stream>>>(pooled, proj_w, proj_b, out);
}
